// MultiHeadAttention_27187142983885
// MI455X (gfx1250) — matrix operand order repaired and checked by reading, not compiled, not run on hardware
//
#include <hip/hip_runtime.h>
#include <math.h>
#include <stdint.h>

// ---------------------------------------------------------------------------
// MI455X (gfx1250) multi-head attention forward.
// bf16 WMMA GEMMs + flash softmax; TDM (tensor_load_to_lds) double-buffered
// K/V staging shared by 2 waves per block; single HBM pass for the attn
// probability output; residual + LayerNorm epilogue in f32.
// ---------------------------------------------------------------------------

typedef __attribute__((ext_vector_type(16))) __bf16        v16bf;
typedef __attribute__((ext_vector_type(8)))  float         v8f;
typedef __attribute__((ext_vector_type(4)))  unsigned int  u32x4;
typedef __attribute__((ext_vector_type(8)))  int           i32x8;
typedef __attribute__((ext_vector_type(4)))  int           i32x4;

#define D_MODEL 1024
#define N_HEAD  16
#define D_HEAD  64
#define BATCH   2
#define T_SEQ   2048
#define M_ROWS  (BATCH * T_SEQ)   // 4096
#define KTILE   32

#if defined(__has_builtin)
#  if __has_builtin(__builtin_amdgcn_tensor_load_to_lds) && \
      __has_builtin(__builtin_amdgcn_s_wait_tensorcnt)
#    define USE_TDM 1
#  endif
#endif
#ifndef USE_TDM
#  define USE_TDM 0
#endif

union BF16Frag { v16bf v; uint4 u[2]; unsigned short h[16]; };
union F8       { v8f   v; float f[8]; };

__device__ inline unsigned short f2bf(float x) {
  union { float f; unsigned int u; } c; c.f = x;
  unsigned int r = c.u + 0x7FFFu + ((c.u >> 16) & 1u);  // RNE
  return (unsigned short)(r >> 16);
}

__device__ inline float redmax16(float x) {
#pragma unroll
  for (int o = 1; o < 16; o <<= 1) x = fmaxf(x, __shfl_xor(x, o, 32));
  return x;
}
__device__ inline float redsum16(float x) {
#pragma unroll
  for (int o = 1; o < 16; o <<= 1) x += __shfl_xor(x, o, 32);
  return x;
}

__device__ inline v8f wmma_bf16(v16bf a, v16bf b, v8f c) {
  return __builtin_amdgcn_wmma_f32_16x16x32_bf16(false, a, false, b,
                                                 (short)0, c, false, false);
}

#if USE_TDM
// 2-D TDM descriptor per CDNA5 ISA.
// data_size = 1 (2-byte elements), D# type = 2 ("image"), groups 2/3/4 zero
// (VADDR2/3 semantics for <=2D tensors), dims/tiles in element units.
// Toolchain builtin is the 6-arg clang-23 form:
//   (u32x4 g0, i32x8 g1, i32x4 g2, i32x4 g3, i32x8 g4, i32 cpol)
__device__ inline void tdm_load_2d(unsigned lds_addr, const void* gptr,
                                   unsigned tensor_d0, unsigned tensor_d1,
                                   unsigned tile_d0, unsigned tile_d1,
                                   unsigned long long stride0) {
  unsigned long long ga = (unsigned long long)(uintptr_t)gptr;
  u32x4 g0;
  g0[0] = 1u;                                   // count=1, user descriptor
  g0[1] = lds_addr;                             // LDS byte address
  g0[2] = (unsigned)(ga & 0xFFFFFFFFu);         // global_addr[31:0]
  g0[3] = (unsigned)((ga >> 32) & 0x01FFFFFFu)  // global_addr[56:32]
          | (2u << 30);                         // type = 2
  i32x8 g1;
  g1[0] = (int)(1u << 16);                      // data_size=1 (2 bytes)
  g1[1] = (int)((tensor_d0 & 0xFFFFu) << 16);   // tensor_dim0[15:0]
  g1[2] = (int)(((tensor_d0 >> 16) & 0xFFFFu) |
                ((tensor_d1 & 0xFFFFu) << 16)); // dim0[31:16] | dim1[15:0]
  g1[3] = (int)(((tensor_d1 >> 16) & 0xFFFFu) |
                ((tile_d0 & 0xFFFFu) << 16));   // dim1[31:16] | tile_dim0
  g1[4] = (int)(tile_d1 & 0xFFFFu);             // tile_dim1 (tile_dim2 = 0)
  g1[5] = (int)(stride0 & 0xFFFFFFFFu);         // tensor_dim0_stride[31:0]
  g1[6] = (int)((stride0 >> 32) & 0xFFFFu);     // stride[47:32], dim1_stride=0
  g1[7] = 0;
  i32x4 z4 = {};
  i32x8 z8 = {};
  __builtin_amdgcn_tensor_load_to_lds(g0, g1, z4, z4, z8, 0);
}
#endif

// ---------------- f32 -> bf16 elementwise --------------------------------
__global__ void cvt_bf16_kernel(const float* __restrict__ src,
                                unsigned short* __restrict__ dst, int n) {
  int i = blockIdx.x * blockDim.x + threadIdx.x;
  if (i < n) dst[i] = f2bf(src[i]);
}

// ---------------- QKV projection: X(4096,1024) x W(1024,1024)^T ----------
// One wave = 16x64 output tile, register-double-buffered over k.
//   transpose_out==0 : dst[bh][t][d]   (Q, K)
//   transpose_out==1 : dst[bh][d][t]   (V, so PV B-frags are contiguous)
__global__ __launch_bounds__(32) void proj_gemm_kernel(
    const unsigned short* __restrict__ X,
    const unsigned short* __restrict__ W,
    const float* __restrict__ bias,
    unsigned short* __restrict__ dst,
    int transpose_out)
{
  const int lane = threadIdx.x;
  const int ln = lane & 15, grp = lane >> 4;
  const int col0 = blockIdx.x * 64;
  const int row0 = blockIdx.y * 16;

  F8 acc[4];
  v8f z = {};
#pragma unroll
  for (int n = 0; n < 4; ++n) acc[n].v = z;

  BF16Frag a0f, a1f, b0f[4], b1f[4];
  const unsigned short* arow = X + (size_t)(row0 + ln) * D_MODEL + grp * 16;

  auto loadA = [&](BF16Frag& a, int k0) {
    const unsigned short* ap = arow + k0;
    a.u[0] = *(const uint4*)ap;
    a.u[1] = *(const uint4*)(ap + 8);
  };
  auto loadB = [&](BF16Frag* bf, int k0) {
#pragma unroll
    for (int n = 0; n < 4; ++n) {
      const unsigned short* bp =
          W + (size_t)(col0 + n * 16 + ln) * D_MODEL + k0 + grp * 16;
      bf[n].u[0] = *(const uint4*)bp;
      bf[n].u[1] = *(const uint4*)(bp + 8);
    }
  };

  loadA(a0f, 0); loadB(b0f, 0);
  for (int k0 = 0; k0 < D_MODEL; k0 += 64) {   // two 32-wide steps / iter
    loadA(a1f, k0 + 32); loadB(b1f, k0 + 32);
#pragma unroll
    for (int n = 0; n < 4; ++n) acc[n].v = wmma_bf16(a0f.v, b0f[n].v, acc[n].v);
    if (k0 + 64 < D_MODEL) { loadA(a0f, k0 + 64); loadB(b0f, k0 + 64); }
#pragma unroll
    for (int n = 0; n < 4; ++n) acc[n].v = wmma_bf16(a1f.v, b1f[n].v, acc[n].v);
  }

#pragma unroll
  for (int n = 0; n < 4; ++n) {
    int col = col0 + n * 16 + ln;
    int h = col >> 6, d = col & 63;
    float bv = bias[col];
#pragma unroll
    for (int j = 0; j < 8; ++j) {
      int row = row0 + j + 8 * grp;             // C layout: row = j + 8*grp
      int b = row >> 11, t = row & (T_SEQ - 1);
      unsigned short val = f2bf(acc[n].f[j] + bv);
      if (transpose_out)
        dst[((size_t)(b * N_HEAD + h) * D_HEAD + d) * T_SEQ + t] = val;
      else
        dst[((size_t)(b * N_HEAD + h) * T_SEQ + t) * D_HEAD + d] = val;
    }
  }
}

// ---------------- flash attention + prob emission ------------------------
// Block = 2 waves sharing TDM-staged K/V tiles; each wave owns one 16-row
// q tile. Sweep 1: online softmax + O accumulation from LDS tiles.
// Sweep 2: recompute scores (L2-resident operands), write normalized probs.
__global__ __launch_bounds__(64) void attention_kernel(
    const unsigned short* __restrict__ qh,    // [B*H][T][64] bf16
    const unsigned short* __restrict__ kh,    // [B*H][T][64] bf16
    const unsigned short* __restrict__ vhT,   // [B*H][64][T] bf16
    const int* __restrict__ mask,             // [B][T][T]
    unsigned short* __restrict__ heads,       // [B][T][H*64] bf16
    float* __restrict__ attn_out)             // [H*B][T][T] f32
{
  __shared__ __align__(16) unsigned short k_lds[2][KTILE * D_HEAD]; // [t][d]
  __shared__ __align__(16) unsigned short v_lds[2][D_HEAD * KTILE]; // [d][t]
  __shared__ __align__(16) unsigned short p_lds[2][16 * 32];

  const int tid = threadIdx.x;
  const int wave = tid >> 5;
  const int lane = tid & 31;
  const int ln = lane & 15, grp = lane >> 4;
  const int h = blockIdx.y, b = blockIdx.z;
  const int bh = b * N_HEAD + h;
  const int qrow0 = (blockIdx.x * 2 + wave) * 16;

  const unsigned short* qp = qh  + (size_t)bh * T_SEQ * D_HEAD;
  const unsigned short* kp = kh  + (size_t)bh * T_SEQ * D_HEAD;
  const unsigned short* vp = vhT + (size_t)bh * D_HEAD * T_SEQ;
  const int* maskb = mask + (size_t)b * T_SEQ * T_SEQ;
  float* attn = attn_out + (size_t)(h * BATCH + b) * T_SEQ * T_SEQ;
  unsigned short* pl = p_lds[wave];

  // Q fragments (d=0..31 and 32..63), live for the whole kernel.
  BF16Frag qf0, qf1;
  {
    const unsigned short* p0 = qp + (size_t)(qrow0 + ln) * D_HEAD + grp * 16;
    qf0.u[0] = *(const uint4*)p0;
    qf0.u[1] = *(const uint4*)(p0 + 8);
    const unsigned short* p1 = p0 + 32;
    qf1.u[0] = *(const uint4*)p1;
    qf1.u[1] = *(const uint4*)(p1 + 8);
  }

  // Stage K tile (KTILE x 64, contiguous rows) and V tile (64 rows of KTILE,
  // row stride T_SEQ) for keys [k0, k0+KTILE) into LDS buffer `buf`.
  auto issue_tiles = [&](int buf, int k0) {
#if USE_TDM
    tdm_load_2d((unsigned)(uintptr_t)(void*)&k_lds[buf][0],
                kp + (size_t)k0 * D_HEAD,
                D_HEAD, T_SEQ, D_HEAD, KTILE, D_HEAD);
    tdm_load_2d((unsigned)(uintptr_t)(void*)&v_lds[buf][0],
                vp + k0,
                T_SEQ, D_HEAD, KTILE, D_HEAD, T_SEQ);
#else
    {  // cooperative copies by the issuing wave (fallback, no TDM builtin)
      const uint4* s = (const uint4*)(kp + (size_t)k0 * D_HEAD);
      uint4* d = (uint4*)&k_lds[buf][0];
      for (int i = lane; i < KTILE * D_HEAD / 8; i += 32) d[i] = s[i];
      uint4* dv = (uint4*)&v_lds[buf][0];
      for (int i = lane; i < D_HEAD * KTILE / 8; i += 32) {
        int r = i >> 2, c = i & 3;
        dv[i] = *(const uint4*)(vp + (size_t)r * T_SEQ + k0 + c * 8);
      }
    }
#endif
  };

  // scores for LDS key sub-tile [n0l, n0l+16): two WMMAs over d
  auto score_tile_lds = [&](const unsigned short* kl, int n0l) -> v8f {
    v8f s = {};
    BF16Frag b0, b1;
    const unsigned short* p0 = kl + (n0l + ln) * D_HEAD + grp * 16;
    b0.u[0] = *(const uint4*)p0;
    b0.u[1] = *(const uint4*)(p0 + 8);
    s = wmma_bf16(qf0.v, b0.v, s);
    const unsigned short* p1 = p0 + 32;
    b1.u[0] = *(const uint4*)p1;
    b1.u[1] = *(const uint4*)(p1 + 8);
    s = wmma_bf16(qf1.v, b1.v, s);
    return s;
  };
  // scores straight from global (sweep 2; operands are L2-resident)
  auto score_tile_glb = [&](int n0) -> v8f {
    v8f s = {};
    BF16Frag b0, b1;
    const unsigned short* p0 = kp + (size_t)(n0 + ln) * D_HEAD + grp * 16;
    b0.u[0] = *(const uint4*)p0;
    b0.u[1] = *(const uint4*)(p0 + 8);
    s = wmma_bf16(qf0.v, b0.v, s);
    const unsigned short* p1 = p0 + 32;
    b1.u[0] = *(const uint4*)p1;
    b1.u[1] = *(const uint4*)(p1 + 8);
    s = wmma_bf16(qf1.v, b1.v, s);
    return s;
  };

  float mrow[8], lrow[8];
#pragma unroll
  for (int j = 0; j < 8; ++j) { mrow[j] = -__builtin_inff(); lrow[j] = 0.f; }
  F8 o[4];
  v8f z = {};
#pragma unroll
  for (int n = 0; n < 4; ++n) o[n].v = z;

  // ---- sweep 1: flash accumulation, TDM double-buffered K/V ----
  if (wave == 0) issue_tiles(0, 0);
  for (int k0 = 0; k0 < T_SEQ; k0 += KTILE) {
    const int cur = (k0 >> 5) & 1;
    const bool has_next = (k0 + KTILE) < T_SEQ;
    if (wave == 0) {
      if (has_next) issue_tiles(cur ^ 1, k0 + KTILE);
#if USE_TDM
      if (has_next) __builtin_amdgcn_s_wait_tensorcnt(2);  // current pair done
      else          __builtin_amdgcn_s_wait_tensorcnt(0);
#endif
    }
    __syncthreads();   // publish LDS tiles to both waves

    const unsigned short* kl = &k_lds[cur][0];
    const unsigned short* vl = &v_lds[cur][0];
    F8 s0, s1;
    s0.v = score_tile_lds(kl, 0);
    s1.v = score_tile_lds(kl, 16);
#pragma unroll
    for (int j = 0; j < 8; ++j) {
      int qg = qrow0 + j + 8 * grp;
      const int* mp = maskb + (size_t)qg * T_SEQ + k0;
      float a0 = s0.f[j] * 0.125f;             // 1/sqrt(64)
      float a1 = s1.f[j] * 0.125f;
      if (mp[ln] == 0)      a0 = -__builtin_inff();
      if (mp[16 + ln] == 0) a1 = -__builtin_inff();
      float tmax = redmax16(fmaxf(a0, a1));
      float mnew = fmaxf(mrow[j], tmax);
      float sc = __expf(mrow[j] - mnew);
      float p0 = __expf(a0 - mnew);
      float p1 = __expf(a1 - mnew);
      lrow[j] = lrow[j] * sc + redsum16(p0 + p1);
      mrow[j] = mnew;
#pragma unroll
      for (int n = 0; n < 4; ++n) o[n].f[j] *= sc;
      pl[(j + 8 * grp) * 32 + ln]      = f2bf(p0);   // C-layout -> LDS
      pl[(j + 8 * grp) * 32 + 16 + ln] = f2bf(p1);
    }
    // reload P as A fragment (in-order DS pipeline orders store->load)
    BF16Frag pa;
    const unsigned short* pp = pl + ln * 32 + grp * 16;
    pa.u[0] = *(const uint4*)pp;
    pa.u[1] = *(const uint4*)(pp + 8);
#pragma unroll
    for (int n = 0; n < 4; ++n) {
      BF16Frag vb;  // B[k][d] = v_lds[d][k], contiguous along k
      const unsigned short* vpp = vl + (n * 16 + ln) * KTILE + grp * 16;
      vb.u[0] = *(const uint4*)vpp;
      vb.u[1] = *(const uint4*)(vpp + 8);
      o[n].v = wmma_bf16(pa.v, vb.v, o[n].v);
    }
    __syncthreads();   // all reads of buffer `cur` done before re-fill
  }

  // ---- normalize O, emit concatenated heads (bf16) ----
#pragma unroll
  for (int n = 0; n < 4; ++n)
#pragma unroll
    for (int j = 0; j < 8; ++j) {
      int t = qrow0 + j + 8 * grp;
      int col = h * D_HEAD + n * 16 + ln;
      heads[((size_t)b * T_SEQ + t) * D_MODEL + col] = f2bf(o[n].f[j] / lrow[j]);
    }

  // ---- sweep 2: recompute scores, single-pass prob write to HBM ----
  float invl[8];
#pragma unroll
  for (int j = 0; j < 8; ++j) invl[j] = 1.0f / lrow[j];

  for (int k0 = 0; k0 < T_SEQ; k0 += KTILE) {
    F8 s0, s1;
    s0.v = score_tile_glb(k0);
    s1.v = score_tile_glb(k0 + 16);
#pragma unroll
    for (int j = 0; j < 8; ++j) {
      int qg = qrow0 + j + 8 * grp;
      const int* mp = maskb + (size_t)qg * T_SEQ + k0;
      float a0 = s0.f[j] * 0.125f;
      float a1 = s1.f[j] * 0.125f;
      if (mp[ln] == 0)      a0 = -__builtin_inff();
      if (mp[16 + ln] == 0) a1 = -__builtin_inff();
      float* ar = attn + (size_t)qg * T_SEQ + k0;
      ar[ln]      = __expf(a0 - mrow[j]) * invl[j];
      ar[16 + ln] = __expf(a1 - mrow[j]) * invl[j];
    }
  }
}

// ---------------- output projection: heads(4096,1024) x Wo^T + bo -> f32 --
__global__ __launch_bounds__(32) void oproj_gemm_kernel(
    const unsigned short* __restrict__ Xh,
    const unsigned short* __restrict__ W,
    const float* __restrict__ bias,
    float* __restrict__ Y)
{
  const int lane = threadIdx.x;
  const int ln = lane & 15, grp = lane >> 4;
  const int col0 = blockIdx.x * 64;
  const int row0 = blockIdx.y * 16;

  F8 acc[4];
  v8f z = {};
#pragma unroll
  for (int n = 0; n < 4; ++n) acc[n].v = z;

  BF16Frag a0f, a1f, b0f[4], b1f[4];
  const unsigned short* arow = Xh + (size_t)(row0 + ln) * D_MODEL + grp * 16;

  auto loadA = [&](BF16Frag& a, int k0) {
    const unsigned short* ap = arow + k0;
    a.u[0] = *(const uint4*)ap;
    a.u[1] = *(const uint4*)(ap + 8);
  };
  auto loadB = [&](BF16Frag* bf, int k0) {
#pragma unroll
    for (int n = 0; n < 4; ++n) {
      const unsigned short* bp =
          W + (size_t)(col0 + n * 16 + ln) * D_MODEL + k0 + grp * 16;
      bf[n].u[0] = *(const uint4*)bp;
      bf[n].u[1] = *(const uint4*)(bp + 8);
    }
  };

  loadA(a0f, 0); loadB(b0f, 0);
  for (int k0 = 0; k0 < D_MODEL; k0 += 64) {
    loadA(a1f, k0 + 32); loadB(b1f, k0 + 32);
#pragma unroll
    for (int n = 0; n < 4; ++n) acc[n].v = wmma_bf16(a0f.v, b0f[n].v, acc[n].v);
    if (k0 + 64 < D_MODEL) { loadA(a0f, k0 + 64); loadB(b0f, k0 + 64); }
#pragma unroll
    for (int n = 0; n < 4; ++n) acc[n].v = wmma_bf16(a1f.v, b1f[n].v, acc[n].v);
  }

#pragma unroll
  for (int n = 0; n < 4; ++n) {
    int col = col0 + n * 16 + ln;
    float bv = bias[col];
#pragma unroll
    for (int j = 0; j < 8; ++j) {
      int row = row0 + j + 8 * grp;
      Y[(size_t)row * D_MODEL + col] = acc[n].f[j] + bv;
    }
  }
}

// ---------------- residual add + LayerNorm -------------------------------
__global__ __launch_bounds__(256) void resid_ln_kernel(
    const float* __restrict__ Y, const float* __restrict__ R,
    const float* __restrict__ g, const float* __restrict__ be,
    float* __restrict__ out)
{
  const int row = blockIdx.x, tid = threadIdx.x;
  __shared__ float s1[256], s2[256];
  const float* yr = Y + (size_t)row * D_MODEL;
  const float* rr = R + (size_t)row * D_MODEL;
  float sum = 0.f, sq = 0.f;
  for (int i = tid; i < D_MODEL; i += 256) {
    float x = yr[i] + rr[i];
    sum += x; sq += x * x;
  }
  s1[tid] = sum; s2[tid] = sq;
  __syncthreads();
  for (int st = 128; st > 0; st >>= 1) {
    if (tid < st) { s1[tid] += s1[tid + st]; s2[tid] += s2[tid + st]; }
    __syncthreads();
  }
  float mu = s1[0] * (1.0f / D_MODEL);
  float var = s2[0] * (1.0f / D_MODEL) - mu * mu;
  float rstd = rsqrtf(var + 1e-5f);
  for (int i = tid; i < D_MODEL; i += 256) {
    float x = yr[i] + rr[i];
    out[(size_t)row * D_MODEL + i] = (x - mu) * rstd * g[i] + be[i];
  }
}

// ---------------------------------------------------------------------------
extern "C" void kernel_launch(void* const* d_in, const int* in_sizes, int n_in,
                              void* d_out, int out_size, void* d_ws, size_t ws_size,
                              hipStream_t stream) {
  (void)in_sizes; (void)n_in; (void)out_size; (void)ws_size;
  const float* q  = (const float*)d_in[0];
  const float* k  = (const float*)d_in[1];
  const float* v  = (const float*)d_in[2];
  const int* mask = (const int*)d_in[3];
  const float* Wq = (const float*)d_in[4];
  const float* bq = (const float*)d_in[5];
  const float* Wk = (const float*)d_in[6];
  const float* bk = (const float*)d_in[7];
  const float* Wv = (const float*)d_in[8];
  const float* bv = (const float*)d_in[9];
  const float* Wo = (const float*)d_in[10];
  const float* bo = (const float*)d_in[11];
  const float* lg = (const float*)d_in[12];
  const float* lb = (const float*)d_in[13];

  char* ws = (char*)d_ws;
  const size_t MB = (size_t)1 << 20;
  unsigned short* XQ    = (unsigned short*)(ws +  0 * MB);  // 8 MB each
  unsigned short* XK    = (unsigned short*)(ws +  8 * MB);
  unsigned short* XV    = (unsigned short*)(ws + 16 * MB);
  unsigned short* WQb   = (unsigned short*)(ws + 24 * MB);  // 2 MB each
  unsigned short* WKb   = (unsigned short*)(ws + 26 * MB);
  unsigned short* WVb   = (unsigned short*)(ws + 28 * MB);
  unsigned short* WOb   = (unsigned short*)(ws + 30 * MB);
  unsigned short* QH    = (unsigned short*)(ws + 32 * MB);  // 8 MB each
  unsigned short* KH    = (unsigned short*)(ws + 40 * MB);
  unsigned short* VHT   = (unsigned short*)(ws + 48 * MB);
  unsigned short* HEADS = (unsigned short*)(ws + 56 * MB);
  float*          Y     = (float*)(ws + 64 * MB);           // 16 MB

  float* outp = (float*)d_out;
  float* attn = outp + (size_t)BATCH * T_SEQ * D_MODEL;

  const int NEL = M_ROWS * D_MODEL;   // 4194304
  const int NW  = D_MODEL * D_MODEL;  // 1048576
  cvt_bf16_kernel<<<(NEL + 255) / 256, 256, 0, stream>>>(q,  XQ,  NEL);
  cvt_bf16_kernel<<<(NEL + 255) / 256, 256, 0, stream>>>(k,  XK,  NEL);
  cvt_bf16_kernel<<<(NEL + 255) / 256, 256, 0, stream>>>(v,  XV,  NEL);
  cvt_bf16_kernel<<<(NW  + 255) / 256, 256, 0, stream>>>(Wq, WQb, NW);
  cvt_bf16_kernel<<<(NW  + 255) / 256, 256, 0, stream>>>(Wk, WKb, NW);
  cvt_bf16_kernel<<<(NW  + 255) / 256, 256, 0, stream>>>(Wv, WVb, NW);
  cvt_bf16_kernel<<<(NW  + 255) / 256, 256, 0, stream>>>(Wo, WOb, NW);

  dim3 pg(D_MODEL / 64, M_ROWS / 16);   // (16, 256) waves of 16x64 tiles
  proj_gemm_kernel<<<pg, 32, 0, stream>>>(XQ, WQb, bq, QH, 0);
  proj_gemm_kernel<<<pg, 32, 0, stream>>>(XK, WKb, bk, KH, 0);
  proj_gemm_kernel<<<pg, 32, 0, stream>>>(XV, WVb, bv, VHT, 1);

  dim3 ag(T_SEQ / KTILE, N_HEAD, BATCH);  // (64, 16, 2) blocks of 2 waves
  attention_kernel<<<ag, 64, 0, stream>>>(QH, KH, VHT, mask, HEADS, attn);

  oproj_gemm_kernel<<<pg, 32, 0, stream>>>(HEADS, WOb, bo, Y);
  resid_ln_kernel<<<M_ROWS, 256, 0, stream>>>(Y, q, lg, lb, outp);
}
